// GGNNConv_48524540510790
// MI455X (gfx1250) — matrix operand order repaired and checked by reading, not compiled, not run on hardware
//
#include <hip/hip_runtime.h>
#include <hip/hip_bf16.h>

typedef __attribute__((ext_vector_type(16))) _Float16 v16h;
typedef __attribute__((ext_vector_type(8)))  _Float16 v8h;
typedef __attribute__((ext_vector_type(8)))  float    v8f;

#define D 128

// ---------------------------------------------------------------------------
// Phase 1: agg[dst] += nodes[src]  (segment_sum over edges)
// One wave per edge -> src/dst wave-uniform; lane moves a float4 and issues
// 4 native GLOBAL_ATOMIC_ADD_F32. nodes (25.6MB) + agg (25.6MB) are L2-resident.
// ---------------------------------------------------------------------------
__global__ __launch_bounds__(256)
void scatter_agg(const float* __restrict__ nodes,
                 const int* __restrict__ adj,   // [2,E]: row0=dst, row1=src
                 float* __restrict__ agg, int E) {
    int tid  = blockIdx.x * blockDim.x + threadIdx.x;
    int e    = tid >> 5;
    if (e >= E) return;
    int lane = tid & 31;
    int dst  = adj[e];
    int src  = adj[E + e];
    const float4* sp = (const float4*)(nodes + (size_t)src * D);
    float4 v = sp[lane];
    float* ap = agg + (size_t)dst * D + lane * 4;
    __hip_atomic_fetch_add(ap + 0, v.x, __ATOMIC_RELAXED, __HIP_MEMORY_SCOPE_AGENT);
    __hip_atomic_fetch_add(ap + 1, v.y, __ATOMIC_RELAXED, __HIP_MEMORY_SCOPE_AGENT);
    __hip_atomic_fetch_add(ap + 2, v.z, __ATOMIC_RELAXED, __HIP_MEMORY_SCOPE_AGENT);
    __hip_atomic_fetch_add(ap + 3, v.w, __ATOMIC_RELAXED, __HIP_MEMORY_SCOPE_AGENT);
}

// ---------------------------------------------------------------------------
// Phase 1b: pack Wr|Wz|Wt (fp32 [256,128] row-major) into f16 fragments laid
// out as V_WMMA_F32_16X16X32_F16 B-operand fragments, k contiguous per lane half (same order as A):
//   frag = mat*64 + nt*8 + kb   (1KB each: 32 lanes x 16 halves)
//   pw[frag*512 + lane*16 + i] = (f16) W[(kb*32 + (lane>>4)*16 + i)*128 + nt*16 + (lane&15)]
// Each GEMM B-fragment then becomes one contiguous 32B read per lane
// (2x global_load_b128) instead of 16 strided b32 loads.
// ---------------------------------------------------------------------------
__global__ __launch_bounds__(256)
void pack_weights(const float* __restrict__ Wr, const float* __restrict__ Wz,
                  const float* __restrict__ Wt, _Float16* __restrict__ pw) {
    int t = blockIdx.x * blockDim.x + threadIdx.x;   // 0..6143
    if (t >= 192 * 32) return;
    int lane = t & 31;
    int frag = t >> 5;                // 0..191
    int kb   = frag & 7;
    int nt   = (frag >> 3) & 7;
    int mat  = frag >> 6;             // 0..2
    const float* W = (mat == 0) ? Wr : (mat == 1) ? Wz : Wt;
    int hs = lane >> 4;
    int nc = nt * 16 + (lane & 15);
    const float* wp = W + (size_t)(kb * 32 + hs * 16) * D + nc;
    _Float16* dst = pw + (size_t)frag * 512 + lane * 16;
#pragma unroll
    for (int i = 0; i < 16; ++i) dst[i] = (_Float16)wp[i * D];
}

// ---------------------------------------------------------------------------
// Phase 2: per 16-node tile (one wave32 each):
//   s = softmax(agg+bias); A = [s | h] (f16 in LDS)
//   r = sigmoid(A@Wr+br), z = sigmoid(A@Wz+bz)      (WMMA f16 -> f32 acc)
//   A2 = [s | r*h]; hhat = tanh(A2@Wt+bt)           (WMMA)
//   out = (1-z)*h + z*hhat
// ---------------------------------------------------------------------------

// A-fragment from LDS row-major A[16][256]:
// lane l: M = l&15, hs = l>>4; halves i=0..15 -> K = 32*kb + 16*hs + i
__device__ inline v16h load_a_frag(const _Float16* A, int m, int hs, int kb) {
    const _Float16* ap = A + m * 256 + kb * 32 + hs * 16;
    v8h lo = *(const v8h*)(ap);
    v8h hi = *(const v8h*)(ap + 8);
    return __builtin_shufflevector(lo, hi, 0,1,2,3,4,5,6,7,8,9,10,11,12,13,14,15);
}

__device__ inline v16h load_b_packed(const _Float16* __restrict__ pw, int frag, int lane) {
    return *(const v16h*)(pw + (size_t)frag * 512 + lane * 16);
}

__device__ inline float sigmoidf_(float x) { return 1.0f / (1.0f + __expf(-x)); }

__global__ __launch_bounds__(64)
void gru_update(const float* __restrict__ nodes,
                const float* __restrict__ agg,
                const float* __restrict__ bias,
                const _Float16* __restrict__ pw,   // packed Wr|Wz|Wt fragments
                const float* __restrict__ br, const float* __restrict__ bz,
                const float* __restrict__ bt,
                float* __restrict__ out, int n) {
    __shared__ _Float16 sA[2][16 * 256];  // [s | h], later [s | r*h]
    __shared__ float    sR[2][16 * D];
    __shared__ float    sZ[2][16 * D];

    int wave = threadIdx.x >> 5;
    int lane = threadIdx.x & 31;
    int tile = blockIdx.x * 2 + wave;
    int row0 = tile * 16;
    if (row0 >= n) return;                 // wave-uniform exit: EXEC stays all-1 for WMMA

    // Warm L2/WGP$ for the packed weights this wave is about to stream (192KB).
    __builtin_prefetch(pw + (size_t)lane * 4096, 0, 1);

    _Float16* A = sA[wave];
    float*    R = sR[wave];
    float*    Z = sZ[wave];

    int m  = lane & 15;                    // row within tile / N-column within D-tile
    int hs = lane >> 4;
    int row = row0 + m;
    int cb  = hs * 64;                     // this lane's 64-column chunk

    // ---- softmax(agg+bias) -> A[:,0:128); stage h -> A[:,128:256) ----
    const float* ag = agg  + (size_t)row * D + cb;
    const float* bp = bias + cb;
    float mx = -3.0e38f;
    for (int j = 0; j < 64; ++j) mx = fmaxf(mx, ag[j] + bp[j]);
    mx = fmaxf(mx, __shfl_xor(mx, 16, 32));
    float sum = 0.0f;
    for (int j = 0; j < 64; ++j) {
        float e = __expf(ag[j] + bp[j] - mx);
        sum += e;
        A[m * 256 + cb + j] = (_Float16)e;
    }
    sum += __shfl_xor(sum, 16, 32);
    float inv = 1.0f / sum;
    const float* hp = nodes + (size_t)row * D + cb;
    for (int j = 0; j < 64; ++j) {
        A[m * 256 + cb + j]       = (_Float16)((float)A[m * 256 + cb + j] * inv);
        A[m * 256 + 128 + cb + j] = (_Float16)hp[j];
    }
    asm volatile("s_wait_dscnt 0" ::: "memory");   // LDS staging visible before fragments load

    // ---- r and z gates: [16,256] @ [256,128], A-fragment shared by both ----
    for (int nt = 0; nt < 8; ++nt) {
        int ncol = nt * 16 + m;
        v8f accR = {}; v8f accZ = {};
#pragma unroll
        for (int kb = 0; kb < 8; ++kb) {
            v16h a  = load_a_frag(A, m, hs, kb);
            v16h bR = load_b_packed(pw, 0 * 64 + nt * 8 + kb, lane);
            v16h bZ = load_b_packed(pw, 1 * 64 + nt * 8 + kb, lane);
            accR = __builtin_amdgcn_wmma_f32_16x16x32_f16(false, a, false, bR, (short)0, accR, false, false);
            accZ = __builtin_amdgcn_wmma_f32_16x16x32_f16(false, a, false, bZ, (short)0, accZ, false, false);
        }
        float brv = br[ncol], bzv = bz[ncol];
#pragma unroll
        for (int j = 0; j < 8; ++j) {              // D layout: VGPR j -> M = j + 8*hs, N = lane&15
            int rr = j + 8 * hs;
            R[rr * D + ncol] = sigmoidf_(accR[j] + brv);
            Z[rr * D + ncol] = sigmoidf_(accZ[j] + bzv);
        }
    }
    asm volatile("s_wait_dscnt 0" ::: "memory");

    // ---- A[:,128:256) = r * h (h currently staged there) ----
    for (int j = 0; j < 64; ++j) {
        int c = cb + j;
        float hv = (float)A[m * 256 + 128 + c];
        A[m * 256 + 128 + c] = (_Float16)(R[m * D + c] * hv);
    }
    asm volatile("s_wait_dscnt 0" ::: "memory");

    // ---- candidate state + GRU blend ----
    for (int nt = 0; nt < 8; ++nt) {
        int ncol = nt * 16 + m;
        v8f acc = {};
#pragma unroll
        for (int kb = 0; kb < 8; ++kb) {
            v16h a  = load_a_frag(A, m, hs, kb);
            v16h bT = load_b_packed(pw, 2 * 64 + nt * 8 + kb, lane);
            acc = __builtin_amdgcn_wmma_f32_16x16x32_f16(false, a, false, bT, (short)0, acc, false, false);
        }
        float btv = bt[ncol];
#pragma unroll
        for (int j = 0; j < 8; ++j) {
            int rr   = j + 8 * hs;
            float hh = tanhf(acc[j] + btv);
            float zz = Z[rr * D + ncol];
            float hv = nodes[(size_t)(row0 + rr) * D + ncol];
            out[(size_t)(row0 + rr) * D + ncol] = (1.0f - zz) * hv + zz * hh;
        }
    }
}

extern "C" void kernel_launch(void* const* d_in, const int* in_sizes, int n_in,
                              void* d_out, int out_size, void* d_ws, size_t ws_size,
                              hipStream_t stream) {
    const float* nodes = (const float*)d_in[0];
    const int*   adj   = (const int*)  d_in[1];
    const float* bias  = (const float*)d_in[2];
    const float* Wr    = (const float*)d_in[3];
    const float* br    = (const float*)d_in[4];
    const float* Wz    = (const float*)d_in[5];
    const float* bz    = (const float*)d_in[6];
    const float* Wt    = (const float*)d_in[7];
    const float* bt    = (const float*)d_in[8];
    float* out = (float*)d_out;

    int n = in_sizes[0] / D;        // 50000
    int E = in_sizes[1] / 2;        // 600000

    // Workspace layout: [ agg: n*128 f32 | pw: 192 fragments * 1KB f16 ]
    float*    agg = (float*)d_ws;
    _Float16* pw  = (_Float16*)((char*)d_ws + (size_t)n * D * sizeof(float));

    hipMemsetAsync(agg, 0, (size_t)n * D * sizeof(float), stream);

    pack_weights<<<(192 * 32 + 255) / 256, 256, 0, stream>>>(Wr, Wz, Wt, pw);

    // All 4 propagation steps are identical (prior_h never updated) -> run once.
    long long sthreads = (long long)E * 32;
    int sblocks = (int)((sthreads + 255) / 256);
    scatter_agg<<<sblocks, 256, 0, stream>>>(nodes, adj, agg, E);

    int ntiles = (n + 15) / 16;
    gru_update<<<(ntiles + 1) / 2, 64, 0, stream>>>(nodes, agg, bias, pw,
                                                    br, bz, bt, out, n);
}
